// ResDCN_89859305767622
// MI455X (gfx1250) — matrix operand order repaired and checked by reading, not compiled, not run on hardware
//
#include <hip/hip_runtime.h>
#include <hip/hip_bf16.h>

typedef _Float16 half_t;
typedef __attribute__((ext_vector_type(16))) _Float16 v16h;
typedef __attribute__((ext_vector_type(8)))  float    v8f;

#define APAD 40   // LDS tile row stride in halves: 80B rows -> b128 stays 16B aligned

union FragU { unsigned int u[8]; v16h h; };

__device__ __forceinline__ float leakyf(float v) { return v > 0.f ? v : 0.01f * v; }

__device__ __forceinline__ unsigned lds_off(const void* p) {
  return (unsigned)(unsigned long long)p;   // generic LDS ptr: low 32 bits = LDS offset
}

__device__ __forceinline__ void async_b128(unsigned ldsoff, const half_t* gaddr) {
  asm volatile("global_load_async_to_lds_b128 %0, %1, off"
               :: "v"(ldsoff), "v"(gaddr) : "memory");
}
__device__ __forceinline__ void wait_async0() {
  asm volatile("s_wait_asynccnt 0x0" ::: "memory");
}

// ---------------------------------------------------------------------------
// small elementwise helpers
// ---------------------------------------------------------------------------
__global__ void __launch_bounds__(256) convert_transpose_f16(
    const float* __restrict__ s,   // K x N row-major f32
    half_t* __restrict__ d,        // N x K row-major f16 (transposed)
    int K, int N) {
  int i = blockIdx.x * blockDim.x + threadIdx.x;
  if (i >= K * N) return;
  int k = i / N, n = i % N;
  d[(size_t)n * K + k] = (half_t)s[i];
}

__global__ void __launch_bounds__(256) zero_f32(float* __restrict__ p, int n) {
  int i = blockIdx.x * blockDim.x + threadIdx.x;
  if (i < n) p[i] = 0.f;
}

// discrete_x[:, :64] -> float into cols 0..63 of x (ld 192), f32 + f16
__global__ void __launch_bounds__(256) embed_kernel(const int* __restrict__ dx,
    float* __restrict__ xF, half_t* __restrict__ xH, int n) {
  int i = blockIdx.x * blockDim.x + threadIdx.x;
  if (i >= n * 64) return;
  int node = i >> 6, c = i & 63;
  float v = (float)dx[(size_t)node * 70 + c];
  xF[(size_t)node * 192 + c] = v;
  xH[(size_t)node * 192 + c] = (half_t)v;
}

// ---------------------------------------------------------------------------
// WMMA GEMM:  out = act( A[M,K](f16) @ B[K,N](f16) + bias + resid )
//  - B supplied PRE-TRANSPOSED: Bt is N x K row-major, ldbt = K
//  - block tile 128(M) x 64(N), 8 waves (4Mx2N), wave tile 32x32 -> 4 WMMA/Kstep
//  - tiles staged via global_load_async_to_lds_b128, double-buffered
//  - N multiple of 64, K multiple of 32 (true for all calls here)
// ---------------------------------------------------------------------------
__global__ void __launch_bounds__(256) gemm_wmma(
    const half_t* __restrict__ A, int lda,
    const half_t* __restrict__ Bt, int ldbt,
    const float*  __restrict__ bias,                  // len N or null
    const float*  __restrict__ resid, int ldr,        // M x N (col-aligned) or null
    float*  __restrict__ outF, int ldoF, int colF,    // may be null
    half_t* __restrict__ outH, int ldoH, int colH,    // may be null
    int Mrows, int K, int Ncols, int act)
{
  __shared__ half_t ldsA[2][128 * APAD];
  __shared__ half_t ldsB[2][64 * APAD];

  const int tid   = threadIdx.x;
  const int lane  = tid & 31;
  const int wid   = tid >> 5;
  const int row0  = blockIdx.y * 128;
  const int c0    = blockIdx.x * 64;
  const int waveM = wid & 3;       // 0..3 -> 32-row slab
  const int waveN = wid >> 2;      // 0..1 -> 32-col slab
  const int lhalf = lane & 15;
  const int hi    = (lane >> 4) & 1;

  // ---- staging map (constant across K steps); clamp rows: no divergent loads
  const int arow0 = tid >> 2;             // 0..63
  const int arow1 = arow0 + 64;           // 64..127
  const int acu   = (tid & 3) * 8;        // half offset within 32-half row
  const int ar0   = min(row0 + arow0, Mrows - 1);
  const int ar1   = min(row0 + arow1, Mrows - 1);
  const half_t* agp0 = A + (size_t)ar0 * lda + acu;
  const half_t* agp1 = A + (size_t)ar1 * lda + acu;
  const half_t* bgp  = Bt + (size_t)(c0 + arow0) * ldbt + acu;   // brow == arow0

  unsigned aoff0[2], aoff1[2], boff[2];
#pragma unroll
  for (int b = 0; b < 2; ++b) {
    aoff0[b] = lds_off(&ldsA[b][arow0 * APAD + acu]);
    aoff1[b] = lds_off(&ldsA[b][arow1 * APAD + acu]);
    boff[b]  = lds_off(&ldsB[b][arow0 * APAD + acu]);
  }

  v8f acc[2][2];
#pragma unroll
  for (int i = 0; i < 2; ++i)
#pragma unroll
    for (int j = 0; j < 2; ++j) acc[i][j] = (v8f){};

  // ---- preload K-step 0 into buffer 0
  async_b128(aoff0[0], agp0);
  async_b128(aoff1[0], agp1);
  async_b128(boff[0],  bgp);
  wait_async0();
  __syncthreads();

  int cur = 0;
  for (int kk = 0; kk < K; kk += 32) {
    // prefetch next K-step into the other buffer (overlaps with WMMA below)
    if (kk + 32 < K) {
      async_b128(aoff0[cur ^ 1], agp0 + kk + 32);
      async_b128(aoff1[cur ^ 1], agp1 + kk + 32);
      async_b128(boff[cur ^ 1],  bgp  + kk + 32);
    }

    // ---- fragment loads per CDNA5 16-bit layouts ----
    FragU fa0, fa1, fb0, fb1;
    {
      const int kbA = hi ? 4 : 0;          // A: lanes>=16 carry K 8-15 / 24-31
      const unsigned int* pa0 = reinterpret_cast<const unsigned int*>(
          &ldsA[cur][(lhalf + waveM * 32) * APAD]);
      const unsigned int* pa1 = reinterpret_cast<const unsigned int*>(
          &ldsA[cur][(lhalf + waveM * 32 + 16) * APAD]);
#pragma unroll
      for (int j = 0; j < 4; ++j) { fa0.u[j] = pa0[kbA + j]; fa0.u[4 + j] = pa0[8 + kbA + j]; }
#pragma unroll
      for (int j = 0; j < 4; ++j) { fa1.u[j] = pa1[kbA + j]; fa1.u[4 + j] = pa1[8 + kbA + j]; }
    }
    {
      const int kbB = hi ? 4 : 0;          // B: lanes>=16 carry K 8-15 / 24-31
      const unsigned int* pb0 = reinterpret_cast<const unsigned int*>(
          &ldsB[cur][(lhalf + waveN * 32) * APAD]);
      const unsigned int* pb1 = reinterpret_cast<const unsigned int*>(
          &ldsB[cur][(lhalf + waveN * 32 + 16) * APAD]);
#pragma unroll
      for (int j = 0; j < 4; ++j) { fb0.u[j] = pb0[kbB + j]; fb0.u[4 + j] = pb0[8 + kbB + j]; fb1.u[j] = pb1[kbB + j]; fb1.u[4 + j] = pb1[8 + kbB + j]; }
    }

    acc[0][0] = __builtin_amdgcn_wmma_f32_16x16x32_f16(false, fa0.h, false, fb0.h,
                                                       (short)0, acc[0][0], false, false);
    acc[0][1] = __builtin_amdgcn_wmma_f32_16x16x32_f16(false, fa0.h, false, fb1.h,
                                                       (short)0, acc[0][1], false, false);
    acc[1][0] = __builtin_amdgcn_wmma_f32_16x16x32_f16(false, fa1.h, false, fb0.h,
                                                       (short)0, acc[1][0], false, false);
    acc[1][1] = __builtin_amdgcn_wmma_f32_16x16x32_f16(false, fa1.h, false, fb1.h,
                                                       (short)0, acc[1][1], false, false);

    wait_async0();       // next tiles landed in LDS
    __syncthreads();     // everyone done reading cur / writing cur^1
    cur ^= 1;
  }

  // ---- epilogue: bias + residual + leaky, dual precision store ----
  const int cb = c0 + waveN * 32;
#pragma unroll
  for (int mt = 0; mt < 2; ++mt) {
#pragma unroll
    for (int nt = 0; nt < 2; ++nt) {
      int gcol = cb + nt * 16 + lhalf;
      float bv = bias ? bias[gcol] : 0.f;
#pragma unroll
      for (int j = 0; j < 8; ++j) {
        int grow = row0 + waveM * 32 + mt * 16 + j + hi * 8;
        if (grow < Mrows) {
          float v = acc[mt][nt][j] + bv;
          if (resid) v += resid[(size_t)grow * ldr + gcol];
          if (act)   v  = leakyf(v);
          if (outF)  outF[(size_t)grow * ldoF + colF + gcol] = v;
          if (outH)  outH[(size_t)grow * ldoH + colH + gcol] = (half_t)v;
        }
      }
    }
  }
}

// ---------------------------------------------------------------------------
// GCN pieces
// ---------------------------------------------------------------------------
__global__ void __launch_bounds__(256) deg_kernel(const int* __restrict__ col,
                                                  float* __restrict__ deg, int ne) {
  int i = blockIdx.x * blockDim.x + threadIdx.x;
  if (i < ne) atomicAdd(&deg[col[i]], 1.f);
}

__global__ void __launch_bounds__(256) dinv_kernel(const float* __restrict__ deg,
                                                   float* __restrict__ dinv, int n) {
  int i = blockIdx.x * blockDim.x + threadIdx.x;
  if (i < n) dinv[i] = rsqrtf(deg[i] + 1.f);
}

// agg = h * dinv^2  (self-loop term)
__global__ void __launch_bounds__(256) agg_init_kernel(const float* __restrict__ h,
    const float* __restrict__ dinv, float* __restrict__ agg, int n) {
  int i = blockIdx.x * blockDim.x + threadIdx.x;
  if (i >= n * 128) return;
  int node = i >> 7;
  float dv = dinv[node];
  agg[i] = h[i] * dv * dv;
}

// agg[col] += h[row] * dinv[row]*dinv[col] ; one edge x 4 dims per thread
__global__ void __launch_bounds__(256) scatter_kernel(const float* __restrict__ h,
    const int* __restrict__ row, const int* __restrict__ col,
    const float* __restrict__ dinv, float* __restrict__ agg, int ne) {
  int idx = blockIdx.x * blockDim.x + threadIdx.x;
  int e = idx >> 5;
  if (e >= ne) return;
  int f0 = (idx & 31) * 4;
  int r = row[e], c = col[e];
  float nrm = dinv[r] * dinv[c];
  const float* hp = h + (size_t)r * 128 + f0;
  float* ap = agg + (size_t)c * 128 + f0;
#pragma unroll
  for (int j = 0; j < 4; ++j) atomicAdd(ap + j, hp[j] * nrm);
}

// out = leaky(agg + b), dual precision with layout params
__global__ void __launch_bounds__(256) gcn_finish_kernel(const float* __restrict__ agg,
    const float* __restrict__ bias,
    float* __restrict__ outF, int ldF, int colF,
    half_t* __restrict__ outH, int ldH, int colH, int n) {
  int i = blockIdx.x * blockDim.x + threadIdx.x;
  if (i >= n * 128) return;
  int node = i >> 7, f = i & 127;
  float v = leakyf(agg[i] + bias[f]);
  outF[(size_t)node * ldF + colF + f] = v;
  outH[(size_t)node * ldH + colH + f] = (half_t)v;
}

// ---------------------------------------------------------------------------
// CrossNet: one wave32 per node, 2 layers in-register, writes z[:,192:384] f16
// ---------------------------------------------------------------------------
__global__ void __launch_bounds__(256) cross_kernel(const float* __restrict__ x,
    const float* __restrict__ cw, const float* __restrict__ cb,
    half_t* __restrict__ zh, int n) {
  int gw = (blockIdx.x * blockDim.x + threadIdx.x) >> 5;
  int lane = threadIdx.x & 31;
  if (gw >= n) return;
  const float* xp = x + (size_t)gw * 192;
  float x0[6], xc[6];
#pragma unroll
  for (int j = 0; j < 6; ++j) { x0[j] = xp[lane + j * 32]; xc[j] = x0[j]; }
#pragma unroll
  for (int i = 0; i < 2; ++i) {
    float p = 0.f;
#pragma unroll
    for (int j = 0; j < 6; ++j) p += xc[j] * cw[i * 192 + lane + j * 32];
    for (int off = 16; off > 0; off >>= 1) p += __shfl_xor(p, off, 32);
#pragma unroll
    for (int j = 0; j < 6; ++j)
      xc[j] = x0[j] * p + cb[i * 192 + lane + j * 32] + xc[j];
  }
  half_t* zp = zh + (size_t)gw * 384 + 192;
#pragma unroll
  for (int j = 0; j < 6; ++j) zp[lane + j * 32] = (half_t)xc[j];
}

// ---------------------------------------------------------------------------
// final layer: y = sigmoid(h[N,320] @ w2[320] + b2); one wave per node
// ---------------------------------------------------------------------------
__global__ void __launch_bounds__(256) pred2_kernel(const half_t* __restrict__ h,
    const float* __restrict__ w2, const float* __restrict__ b2,
    float* __restrict__ out, int n) {
  int gw = (blockIdx.x * blockDim.x + threadIdx.x) >> 5;
  int lane = threadIdx.x & 31;
  if (gw >= n) return;
  const half_t* hp = h + (size_t)gw * 320;
  float p = 0.f;
#pragma unroll
  for (int j = 0; j < 10; ++j) {
    int f = lane + j * 32;
    p += (float)hp[f] * w2[f];
  }
  for (int off = 16; off > 0; off >>= 1) p += __shfl_xor(p, off, 32);
  if (lane == 0) out[gw] = 1.f / (1.f + expf(-(p + b2[0])));
}

// ---------------------------------------------------------------------------
// host side
// ---------------------------------------------------------------------------
static inline size_t align_up(size_t x) { return (x + 255) & ~(size_t)255; }

extern "C" void kernel_launch(void* const* d_in, const int* in_sizes, int n_in,
                              void* d_out, int out_size, void* d_ws, size_t ws_size,
                              hipStream_t stream) {
  const int n  = in_sizes[0] / 70;   // 50000
  const int ne = in_sizes[1] / 2;    // 600000

  const int*   dx       = (const int*)d_in[0];
  const int*   e_row    = (const int*)d_in[1];
  const int*   e_col    = e_row + ne;
  const float* w_g0     = (const float*)d_in[2];
  const float* b_g0     = (const float*)d_in[3];
  const float* w_gcn1   = (const float*)d_in[4];
  const float* b_gcn1   = (const float*)d_in[5];
  const float* w_gcn2   = (const float*)d_in[6];
  const float* b_gcn2   = (const float*)d_in[7];
  const float* res_w1   = (const float*)d_in[8];   // (2,192,512)
  const float* res_b1   = (const float*)d_in[9];   // (2,512)
  const float* res_w2   = (const float*)d_in[10];  // (2,512,192)
  const float* res_b2   = (const float*)d_in[11];  // (2,192)
  const float* cross_w  = (const float*)d_in[12];  // (2,192)
  const float* cross_b  = (const float*)d_in[13];
  const float* p_w1     = (const float*)d_in[14];  // (384,320)
  const float* p_b1     = (const float*)d_in[15];
  const float* p_w2     = (const float*)d_in[16];  // (320,1)
  const float* p_b2     = (const float*)d_in[17];
  float* out = (float*)d_out;

  char* ws = (char*)d_ws;
  size_t o = 0;
  // f16 transposed weight copies (N x K each)
  const int cW0 = 64 * 128, cW1 = 128 * 128, cW2 = 128 * 128;
  const int cR1 = 192 * 512, cR2 = 512 * 192, cP1 = 384 * 320;
  size_t oW0 = o; o = align_up(o + (size_t)cW0 * 2);
  size_t oW1 = o; o = align_up(o + (size_t)cW1 * 2);
  size_t oW2 = o; o = align_up(o + (size_t)cW2 * 2);
  size_t oR1 = o; o = align_up(o + (size_t)cR1 * 2 * 2);   // 2 layers
  size_t oR2 = o; o = align_up(o + (size_t)cR2 * 2 * 2);   // 2 layers
  size_t oP1 = o; o = align_up(o + (size_t)cP1 * 2);
  // persistent activations
  size_t oXF = o; o = align_up(o + (size_t)n * 192 * 4);   // x (f32)
  size_t oXH = o; o = align_up(o + (size_t)n * 192 * 2);   // x (f16)
  // scratch region: GCN phase and ResDNN/pred phase overlap
  size_t sb = o;
  size_t p = sb;
  size_t oXG  = p; p = align_up(p + (size_t)n * 128 * 4);  // x_g f32
  size_t oXGH = p; p = align_up(p + (size_t)n * 128 * 2);  // x_g f16
  size_t oHB  = p; p = align_up(p + (size_t)n * 128 * 4);  // gcn linear h
  size_t oAG  = p; p = align_up(p + (size_t)n * 128 * 4);  // agg
  size_t oDEG = p; p = align_up(p + (size_t)n * 4);
  size_t oDIV = p; p = align_up(p + (size_t)n * 4);
  size_t q = sb;                                           // overlapped
  size_t oHR  = q; q = align_up(q + (size_t)n * 512 * 2);  // res hidden f16
  size_t oD1F = q; q = align_up(q + (size_t)n * 192 * 4);  // x_deep f32
  size_t oD1H = q; q = align_up(q + (size_t)n * 192 * 2);  // x_deep f16
  size_t oZH  = q; q = align_up(q + (size_t)n * 384 * 2);  // z f16
  size_t oPH  = q; q = align_up(q + (size_t)n * 320 * 2);  // pred hidden f16
  (void)ws_size;

  half_t* wg0t = (half_t*)(ws + oW0);
  half_t* wg1t = (half_t*)(ws + oW1);
  half_t* wg2t = (half_t*)(ws + oW2);
  half_t* rw1t = (half_t*)(ws + oR1);
  half_t* rw2t = (half_t*)(ws + oR2);
  half_t* pw1t = (half_t*)(ws + oP1);
  float*  xF   = (float*)(ws + oXF);
  half_t* xH   = (half_t*)(ws + oXH);
  float*  xg   = (float*)(ws + oXG);
  half_t* xgh  = (half_t*)(ws + oXGH);
  float*  hb   = (float*)(ws + oHB);
  float*  agg  = (float*)(ws + oAG);
  float*  deg  = (float*)(ws + oDEG);
  float*  dinv = (float*)(ws + oDIV);
  half_t* hrh  = (half_t*)(ws + oHR);
  float*  xd1f = (float*)(ws + oD1F);
  half_t* xd1h = (half_t*)(ws + oD1H);
  half_t* zh   = (half_t*)(ws + oZH);
  half_t* ph   = (half_t*)(ws + oPH);

  const int TB = 256;
  #define GRID1(cnt) dim3((unsigned)(((size_t)(cnt) + TB - 1) / TB))

  // --- weight conversion + transpose (once) ---
  convert_transpose_f16<<<GRID1(cW0), TB, 0, stream>>>(w_g0,   wg0t, 64, 128);
  convert_transpose_f16<<<GRID1(cW1), TB, 0, stream>>>(w_gcn1, wg1t, 128, 128);
  convert_transpose_f16<<<GRID1(cW2), TB, 0, stream>>>(w_gcn2, wg2t, 128, 128);
  for (int l = 0; l < 2; ++l) {
    convert_transpose_f16<<<GRID1(cR1), TB, 0, stream>>>(res_w1 + (size_t)l * cR1,
                                                         rw1t + (size_t)l * cR1, 192, 512);
    convert_transpose_f16<<<GRID1(cR2), TB, 0, stream>>>(res_w2 + (size_t)l * cR2,
                                                         rw2t + (size_t)l * cR2, 512, 192);
  }
  convert_transpose_f16<<<GRID1(cP1), TB, 0, stream>>>(p_w1, pw1t, 384, 320);

  // --- embedding of discrete features into x[:, :64] ---
  embed_kernel<<<GRID1(n * 64), TB, 0, stream>>>(dx, xF, xH, n);

  // --- degrees / norms (shared by both GCN layers) ---
  zero_f32<<<GRID1(n), TB, 0, stream>>>(deg, n);
  deg_kernel<<<GRID1(ne), TB, 0, stream>>>(e_col, deg, ne);
  dinv_kernel<<<GRID1(n), TB, 0, stream>>>(deg, dinv, n);

  const unsigned gy = (unsigned)((n + 127) / 128);

  // --- x_g0 = leaky(x_d @ w_g0 + b_g0) ---
  gemm_wmma<<<dim3(128 / 64, gy), TB, 0, stream>>>(xH, 192, wg0t, 64, b_g0, nullptr, 0,
                                                   xg, 128, 0, xgh, 128, 0, n, 64, 128, 1);

  // --- 2 GCN layers ---
  for (int layer = 0; layer < 2; ++layer) {
    half_t* W = layer ? wg2t : wg1t;
    const float* bb = layer ? b_gcn2 : b_gcn1;
    gemm_wmma<<<dim3(128 / 64, gy), TB, 0, stream>>>(xgh, 128, W, 128, nullptr, nullptr, 0,
                                                     hb, 128, 0, nullptr, 0, 0, n, 128, 128, 0);
    agg_init_kernel<<<GRID1(n * 128), TB, 0, stream>>>(hb, dinv, agg, n);
    scatter_kernel<<<GRID1((size_t)ne * 32), TB, 0, stream>>>(hb, e_row, e_col, dinv, agg, ne);
    if (layer == 0)
      gcn_finish_kernel<<<GRID1(n * 128), TB, 0, stream>>>(agg, bb, xg, 128, 0,
                                                           xgh, 128, 0, n);
    else  // final GCN output -> x[:, 64:192]
      gcn_finish_kernel<<<GRID1(n * 128), TB, 0, stream>>>(agg, bb, xF, 192, 64,
                                                           xH, 192, 64, n);
  }

  // --- ResDNN block 0: x -> hid(512) -> x_deep1 ---
  gemm_wmma<<<dim3(512 / 64, gy), TB, 0, stream>>>(xH, 192, rw1t, 192, res_b1, nullptr, 0,
                                                   nullptr, 0, 0, hrh, 512, 0, n, 192, 512, 1);
  gemm_wmma<<<dim3(192 / 64, gy), TB, 0, stream>>>(hrh, 512, rw2t, 512, res_b2, xF, 192,
                                                   xd1f, 192, 0, xd1h, 192, 0, n, 512, 192, 1);
  // --- ResDNN block 1: x_deep1 -> hid -> z[:, :192] ---
  gemm_wmma<<<dim3(512 / 64, gy), TB, 0, stream>>>(xd1h, 192, rw1t + (size_t)cR1, 192,
                                                   res_b1 + 512, nullptr, 0,
                                                   nullptr, 0, 0, hrh, 512, 0, n, 192, 512, 1);
  gemm_wmma<<<dim3(192 / 64, gy), TB, 0, stream>>>(hrh, 512, rw2t + (size_t)cR2, 512,
                                                   res_b2 + 192, xd1f, 192,
                                                   nullptr, 0, 0, zh, 384, 0, n, 512, 192, 1);

  // --- CrossNet -> z[:, 192:384] ---
  cross_kernel<<<GRID1((size_t)n * 32), TB, 0, stream>>>(xF, cross_w, cross_b, zh, n);

  // --- prediction MLP layer 1: z @ p_w1 -> leaky ---
  gemm_wmma<<<dim3(320 / 64, gy), TB, 0, stream>>>(zh, 384, pw1t, 384, p_b1, nullptr, 0,
                                                   nullptr, 0, 0, ph, 320, 0, n, 384, 320, 1);
  // --- layer 2 + sigmoid ---
  pred2_kernel<<<GRID1((size_t)n * 32), TB, 0, stream>>>(ph, p_w2, p_b2, out, n);
  (void)out_size; (void)n_in;
}
